// LATTICEModel_18210661335606
// MI455X (gfx1250) — matrix operand order repaired and checked by reading, not compiled, not run on hardware
//
#include <hip/hip_runtime.h>
#include <hip/hip_bf16.h>

typedef __attribute__((ext_vector_type(16))) _Float16 v16h;
typedef __attribute__((ext_vector_type(8)))  _Float16 v8h;
typedef __attribute__((ext_vector_type(8)))  float    v8f;

#define U_N   16384
#define I_N   8192
#define NN_N  (U_N + I_N)
#define KE    64
#define D0_N  2048
#define D1_N  512
#define TOPK  10
#define LM    0.7f

// ---------------- tiny scalar kernels ----------------

__global__ void k_softmax2(const float* w, float* sm) {
    if (blockIdx.x == 0 && threadIdx.x == 0) {
        float a = w[0], b = w[1];
        float m = fmaxf(a, b);
        float ea = expf(a - m), eb = expf(b - m);
        float inv = 1.0f / (ea + eb);
        sm[0] = ea * inv; sm[1] = eb * inv;
    }
}

__global__ void k_cvt_f16(const float* in, _Float16* out, long long n) {
    long long t = (long long)blockIdx.x * blockDim.x + threadIdx.x;
    if (t < n) out[t] = (_Float16)in[t];
}

// transpose W (Kd x 64, f32) -> Wt (64 x Kd, f16) so GEMM B-fragments are contiguous
__global__ void k_transpose_w_f16(const float* __restrict__ W, _Float16* __restrict__ Wt, int Kd) {
    long long t = (long long)blockIdx.x * blockDim.x + threadIdx.x;
    if (t >= (long long)Kd * KE) return;
    int k = (int)(t >> 6);      // 0..Kd-1
    int n = (int)(t & 63);      // 0..63
    Wt[(size_t)n * Kd + k] = (_Float16)W[(size_t)k * KE + n];
}

__global__ void k_fill_f32(float* p, float v, long long n) {
    long long t = (long long)blockIdx.x * blockDim.x + threadIdx.x;
    if (t < n) p[t] = v;
}

__global__ void k_recip(float* p, long long n) {
    long long t = (long long)blockIdx.x * blockDim.x + threadIdx.x;
    if (t < n) p[t] = 1.0f / p[t];
}

// ---------------- WMMA GEMM: C[MxNc] = A[MxKd](f16) @ Bt[NcxKd](f16)^T + bias ----------------
// Bt is N-major (already transposed), so every lane's B fragment is one aligned v16h load.
// one wave computes one 16x16 output tile.

__global__ __launch_bounds__(256) void k_gemm_f16t(const _Float16* __restrict__ A,
                                                   const _Float16* __restrict__ Bt,
                                                   const float* __restrict__ bias,
                                                   float* __restrict__ C,
                                                   int M, int Nc, int Kd) {
    int wave = (int)((blockIdx.x * blockDim.x + threadIdx.x) >> 5);
    int lane = threadIdx.x & 31;
    int tilesN = Nc >> 4;
    int tm = wave / tilesN, tn = wave % tilesN;
    if (tm * 16 >= M) return;            // wave-uniform guard
    int m    = tm * 16 + (lane & 15);
    int n    = tn * 16 + (lane & 15);
    int koff = (lane & 16) ? 16 : 0;     // A: lanes 0-15 hold K 0..15, 16-31 hold K 16..31
    int kb2  = (lane & 16) ? 16 : 0;     // B: lanes 0-15 hold K 0..15, 16-31 hold K 16..31
    v8f c = {};
    for (int kb = 0; kb < Kd; kb += 32) {
        v8h a0 = *(const v8h*)(A + (size_t)m * Kd + kb + koff);
        v8h a1 = *(const v8h*)(A + (size_t)m * Kd + kb + koff + 8);
        v16h a;
#pragma unroll
        for (int j = 0; j < 8; ++j) { a[j] = a0[j]; a[8 + j] = a1[j]; }
        v16h b = *(const v16h*)(Bt + (size_t)n * Kd + kb + kb2);
        c = __builtin_amdgcn_wmma_f32_16x16x32_f16(false, a, false, b, (short)0, c, false, false);
    }
    float bv = bias ? bias[n] : 0.0f;
    int rbase = tm * 16 + ((lane & 16) ? 8 : 0);   // D: lanes 0-15 -> M=r, lanes 16-31 -> M=8+r
#pragma unroll
    for (int r = 0; r < 8; ++r) C[(size_t)(rbase + r) * Nc + n] = c[r] + bv;
}

// ---------------- row normalization ----------------

__global__ __launch_bounds__(64) void k_rownorm_f32_to_f16(const float* __restrict__ X,
                                                           _Float16* __restrict__ Y, int C) {
    __shared__ float red[64];
    int r = blockIdx.x;
    float s = 0.0f;
    for (int j = threadIdx.x; j < C; j += 64) { float v = X[(size_t)r * C + j]; s += v * v; }
    red[threadIdx.x] = s; __syncthreads();
    for (int st = 32; st > 0; st >>= 1) {
        if (threadIdx.x < st) red[threadIdx.x] += red[threadIdx.x + st];
        __syncthreads();
    }
    float inv = 1.0f / fmaxf(sqrtf(red[0]), 1e-30f);
    for (int j = threadIdx.x; j < C; j += 64)
        Y[(size_t)r * C + j] = (_Float16)(X[(size_t)r * C + j] * inv);
}

__global__ __launch_bounds__(256) void k_rownorm_f16_inplace(_Float16* __restrict__ X, int C) {
    __shared__ float red[256];
    int r = blockIdx.x;
    _Float16* row = X + (size_t)r * C;
    float s = 0.0f;
    for (int j = threadIdx.x; j < C; j += 256) { float v = (float)row[j]; s += v * v; }
    red[threadIdx.x] = s; __syncthreads();
    for (int st = 128; st > 0; st >>= 1) {
        if (threadIdx.x < st) red[threadIdx.x] += red[threadIdx.x + st];
        __syncthreads();
    }
    float inv = 1.0f / fmaxf(sqrtf(red[0]), 1e-30f);
    __syncthreads();
    for (int j = threadIdx.x; j < C; j += 256) row[j] = (_Float16)((float)row[j] * inv);
}

// ---------------- fused cosine-sim GEMM + per-row top-10 ----------------
// block = 256 thr = 8 waves; handles 16 rows; sweeps all 8192 columns in 128-col chunks.
// sim tile lives only in registers/LDS -> no 256MB sim matrix in HBM.

__global__ __launch_bounds__(256) void k_sim_topk(const _Float16* __restrict__ F, int Dd,
                                                  int* __restrict__ knn_idx,
                                                  float* __restrict__ knn_val) {
    __shared__ float tile[16][132];
    int rbase = blockIdx.x * 16;
    int wv   = threadIdx.x >> 5;
    int lane = threadIdx.x & 31;
    int tid  = threadIdx.x;

    float tv[TOPK]; int tc[TOPK];
#pragma unroll
    for (int j = 0; j < TOPK; ++j) { tv[j] = -3.0e38f; tc[j] = 0; }

    int mrow = rbase + (lane & 15);
    int koff = (lane & 16) ? 16 : 0;
    int kb2  = (lane & 16) ? 16 : 0;

    for (int ct = 0; ct < I_N / 128; ++ct) {
        int n0   = ct * 128 + wv * 16;
        int ncol = n0 + (lane & 15);
        v8f c = {};
        for (int kb = 0; kb < Dd; kb += 32) {
            v8h a0 = *(const v8h*)(F + (size_t)mrow * Dd + kb + koff);
            v8h a1 = *(const v8h*)(F + (size_t)mrow * Dd + kb + koff + 8);
            v16h a;
#pragma unroll
            for (int j = 0; j < 8; ++j) { a[j] = a0[j]; a[8 + j] = a1[j]; }
            v16h b = *(const v16h*)(F + (size_t)ncol * Dd + kb + kb2);
            if (kb + 32 < Dd)
                __builtin_prefetch((const void*)(F + (size_t)ncol * Dd + kb + 32 + kb2), 0, 3);
            c = __builtin_amdgcn_wmma_f32_16x16x32_f16(false, a, false, b, (short)0, c, false, false);
        }
        int cl = wv * 16 + (lane & 15);
        int rb = (lane & 16) ? 8 : 0;
#pragma unroll
        for (int r = 0; r < 8; ++r) tile[rb + r][cl] = c[r];
        __syncthreads();
        if (tid < 16) {
            for (int j = 0; j < 128; ++j) {
                float v = tile[tid][j];
                if (v > tv[TOPK - 1]) {
                    int col = ct * 128 + j;
                    int p = TOPK - 1;
                    while (p > 0 && tv[p - 1] < v) { tv[p] = tv[p - 1]; tc[p] = tc[p - 1]; --p; }
                    tv[p] = v; tc[p] = col;
                }
            }
        }
        __syncthreads();
    }
    if (tid < 16) {
        int r = rbase + tid;
#pragma unroll
        for (int j = 0; j < TOPK; ++j) {
            knn_idx[(size_t)r * TOPK + j] = tc[j];
            knn_val[(size_t)r * TOPK + j] = tv[j];
        }
    }
}

// ---------------- item-graph laplacians (rows are repeat(arange,k): per-row, no atomics) ----------------

__global__ void k_item_dinv(const int* iL0, const float* vL0, const int* iL1, const float* vL1,
                            const int* iO0, const float* vO0, const int* iO1, const float* vO1,
                            const float* smw, float* dinv_l, float* dinv_o0, float* dinv_o1) {
    int r = blockIdx.x * blockDim.x + threadIdx.x;
    if (r >= I_N) return;
    float s0 = smw[0], s1 = smw[1];
    float dl = 1.0f, d0 = 1.0f, d1 = 1.0f;   // +1 from fill_diag identity
    for (int j = 0; j < TOPK; ++j) {
        int c0 = iL0[r * TOPK + j]; if (c0 != r) dl += vL0[r * TOPK + j] * s0;
        int c1 = iL1[r * TOPK + j]; if (c1 != r) dl += vL1[r * TOPK + j] * s1;
        int c2 = iO0[r * TOPK + j]; if (c2 != r) d0 += vO0[r * TOPK + j];
        int c3 = iO1[r * TOPK + j]; if (c3 != r) d1 += vO1[r * TOPK + j];
    }
    dinv_l[r]  = (dl == 0.0f) ? 0.0f : rsqrtf(dl);
    dinv_o0[r] = (d0 == 0.0f) ? 0.0f : rsqrtf(d0);
    dinv_o1[r] = (d1 == 0.0f) ? 0.0f : rsqrtf(d1);
}

__global__ void k_item_spmm(const float* __restrict__ gi,
                            const int* iL0, const float* vL0, const int* iL1, const float* vL1,
                            const int* iO0, const float* vO0, const int* iO1, const float* vO1,
                            const float* smw, const float* dinv_l, const float* dinv_o0,
                            const float* dinv_o1, float* __restrict__ item_emb) {
    int t = blockIdx.x * blockDim.x + threadIdx.x;
    int r = t >> 6, k = t & 63;
    if (r >= I_N) return;
    float s0 = smw[0], s1 = smw[1];
    float dlr = dinv_l[r], d0r = dinv_o0[r], d1r = dinv_o1[r];
    float acc = ((1.0f - LM) * dlr * dlr + LM * (s0 * d0r * d0r + s1 * d1r * d1r))
                * gi[(size_t)r * KE + k];
    for (int j = 0; j < TOPK; ++j) {
        int c;
        c = iL0[r * TOPK + j];
        if (c != r) acc += (1.0f - LM) * vL0[r * TOPK + j] * s0 * dlr * dinv_l[c] * gi[(size_t)c * KE + k];
        c = iL1[r * TOPK + j];
        if (c != r) acc += (1.0f - LM) * vL1[r * TOPK + j] * s1 * dlr * dinv_l[c] * gi[(size_t)c * KE + k];
        c = iO0[r * TOPK + j];
        if (c != r) acc += LM * s0 * vO0[r * TOPK + j] * d0r * dinv_o0[c] * gi[(size_t)c * KE + k];
        c = iO1[r * TOPK + j];
        if (c != r) acc += LM * s1 * vO1[r * TOPK + j] * d1r * dinv_o1[c] * gi[(size_t)c * KE + k];
    }
    item_emb[(size_t)r * KE + k] = acc;
}

// ---------------- user-item bipartite graph ----------------

__global__ void k_ui_deg(const int* src, const int* dst, float* deg, int E) {
    int e = blockIdx.x * blockDim.x + threadIdx.x;
    if (e >= E) return;
    atomicAdd(&deg[src[e]], 1.0f);
    atomicAdd(&deg[U_N + dst[e]], 1.0f);
}

__global__ void k_concat_ego(const float* gu, const float* gi, float* e0) {
    long long t = (long long)blockIdx.x * blockDim.x + threadIdx.x;
    if (t >= (long long)NN_N * KE) return;
    e0[t] = (t < (long long)U_N * KE) ? gu[t] : gi[t - (long long)U_N * KE];
}

__global__ void k_ui_diag(const float* __restrict__ dinv, const float* __restrict__ x,
                          float* __restrict__ y) {
    long long t = (long long)blockIdx.x * blockDim.x + threadIdx.x;
    if (t >= (long long)NN_N * KE) return;
    int n = (int)(t >> 6);
    float d = dinv[n];
    y[t] = d * d * x[t];
}

__global__ void k_ui_edges(const int* __restrict__ src, const int* __restrict__ dst,
                           const float* __restrict__ dinv, const float* __restrict__ x,
                           float* __restrict__ y, int E) {
    long long t = (long long)blockIdx.x * blockDim.x + threadIdx.x;
    if (t >= (long long)E * KE) return;
    int e = (int)(t >> 6), k = (int)(t & 63);
    int s = src[e], d = U_N + dst[e];
    float w = dinv[s] * dinv[d];
    atomicAdd(&y[(size_t)s * KE + k], w * x[(size_t)d * KE + k]);
    atomicAdd(&y[(size_t)d * KE + k], w * x[(size_t)s * KE + k]);
}

__global__ __launch_bounds__(64) void k_l2norm_rows(float* __restrict__ x) {
    __shared__ float red[64];
    size_t base = (size_t)blockIdx.x * KE;
    float v = x[base + threadIdx.x];
    red[threadIdx.x] = v * v; __syncthreads();
    for (int st = 32; st > 0; st >>= 1) {
        if (threadIdx.x < st) red[threadIdx.x] += red[threadIdx.x + st];
        __syncthreads();
    }
    float inv = 1.0f / fmaxf(sqrtf(red[0]), 1e-12f);
    x[base + threadIdx.x] = v * inv;
}

__global__ void k_final_u(const float* e0, const float* e1, const float* e2, float* out) {
    long long t = (long long)blockIdx.x * blockDim.x + threadIdx.x;
    if (t >= (long long)U_N * KE) return;
    out[t] = (e0[t] + e1[t] + e2[t]) * (1.0f / 3.0f);
}

__global__ __launch_bounds__(64) void k_final_i(const float* e0, const float* e1, const float* e2,
                                                const float* item_emb, float* out) {
    __shared__ float red[64];
    int r = blockIdx.x, k = threadIdx.x;
    size_t off = (size_t)(U_N + r) * KE + k;
    float mean = (e0[off] + e1[off] + e2[off]) * (1.0f / 3.0f);
    float v = item_emb[(size_t)r * KE + k];
    red[k] = v * v; __syncthreads();
    for (int st = 32; st > 0; st >>= 1) {
        if (k < st) red[k] += red[k + st];
        __syncthreads();
    }
    float inv = 1.0f / fmaxf(sqrtf(red[0]), 1e-12f);
    out[(size_t)U_N * KE + (size_t)r * KE + k] = mean + v * inv;
}

// ---------------- host launcher ----------------

extern "C" void kernel_launch(void* const* d_in, const int* in_sizes, int n_in,
                              void* d_out, int out_size, void* d_ws, size_t ws_size,
                              hipStream_t stream) {
    const float* gu    = (const float*)d_in[0];
    const float* gi    = (const float*)d_in[1];
    const float* feat0 = (const float*)d_in[2];
    const float* feat1 = (const float*)d_in[3];
    const float* w0    = (const float*)d_in[4];
    const float* b0    = (const float*)d_in[5];
    const float* w1    = (const float*)d_in[6];
    const float* b1    = (const float*)d_in[7];
    const float* imp_w = (const float*)d_in[8];
    const int* ui_src  = (const int*)d_in[9];
    const int* ui_dst  = (const int*)d_in[10];
    const int E = in_sizes[9];
    float* out = (float*)d_out;

    char* ws = (char*)d_ws;
    size_t off = 0;
    auto alloc = [&](size_t bytes) -> void* {
        void* p = ws + off;
        off = (off + bytes + 255) & ~(size_t)255;
        return p;
    };

    float*    sm_w  = (float*)alloc(2 * sizeof(float));
    _Float16* f0h   = (_Float16*)alloc((size_t)I_N * D0_N * 2);   // raw f16 -> normalized in place later
    _Float16* f1h   = (_Float16*)alloc((size_t)I_N * D1_N * 2);
    _Float16* w0th  = (_Float16*)alloc((size_t)D0_N * KE * 2);    // transposed weights (64 x Kd)
    _Float16* w1th  = (_Float16*)alloc((size_t)D1_N * KE * 2);
    float*    p0    = (float*)alloc((size_t)I_N * KE * 4);
    float*    p1    = (float*)alloc((size_t)I_N * KE * 4);
    _Float16* pn0h  = (_Float16*)alloc((size_t)I_N * KE * 2);
    _Float16* pn1h  = (_Float16*)alloc((size_t)I_N * KE * 2);
    int*   iL0 = (int*)alloc((size_t)I_N * TOPK * 4);
    float* vL0 = (float*)alloc((size_t)I_N * TOPK * 4);
    int*   iL1 = (int*)alloc((size_t)I_N * TOPK * 4);
    float* vL1 = (float*)alloc((size_t)I_N * TOPK * 4);
    int*   iO0 = (int*)alloc((size_t)I_N * TOPK * 4);
    float* vO0 = (float*)alloc((size_t)I_N * TOPK * 4);
    int*   iO1 = (int*)alloc((size_t)I_N * TOPK * 4);
    float* vO1 = (float*)alloc((size_t)I_N * TOPK * 4);
    float* dinv_l  = (float*)alloc((size_t)I_N * 4);
    float* dinv_o0 = (float*)alloc((size_t)I_N * 4);
    float* dinv_o1 = (float*)alloc((size_t)I_N * 4);
    float* dinv_a  = (float*)alloc((size_t)NN_N * 4);
    float* item_emb = (float*)alloc((size_t)I_N * KE * 4);
    float* e0 = (float*)alloc((size_t)NN_N * KE * 4);
    float* e1 = (float*)alloc((size_t)NN_N * KE * 4);
    float* e2 = (float*)alloc((size_t)NN_N * KE * 4);
    (void)ws_size; (void)n_in; (void)out_size;

    auto cdiv = [](long long a, long long b) -> int { return (int)((a + b - 1) / b); };

    // softmax over importance weights
    k_softmax2<<<1, 32, 0, stream>>>(imp_w, sm_w);

    // f32 -> f16 conversions + weight transposes
    k_cvt_f16<<<cdiv((long long)I_N * D0_N, 256), 256, 0, stream>>>(feat0, f0h, (long long)I_N * D0_N);
    k_cvt_f16<<<cdiv((long long)I_N * D1_N, 256), 256, 0, stream>>>(feat1, f1h, (long long)I_N * D1_N);
    k_transpose_w_f16<<<cdiv((long long)D0_N * KE, 256), 256, 0, stream>>>(w0, w0th, D0_N);
    k_transpose_w_f16<<<cdiv((long long)D1_N * KE, 256), 256, 0, stream>>>(w1, w1th, D1_N);

    // projections p_m = feat_m @ w_m + b_m   (WMMA, contiguous B via transposed weights)
    {
        int waves = (I_N / 16) * (KE / 16);
        k_gemm_f16t<<<cdiv((long long)waves * 32, 256), 256, 0, stream>>>(f0h, w0th, b0, p0, I_N, KE, D0_N);
        k_gemm_f16t<<<cdiv((long long)waves * 32, 256), 256, 0, stream>>>(f1h, w1th, b1, p1, I_N, KE, D1_N);
    }

    // normalized f16 copies for cosine similarity
    k_rownorm_f32_to_f16<<<I_N, 64, 0, stream>>>(p0, pn0h, KE);
    k_rownorm_f32_to_f16<<<I_N, 64, 0, stream>>>(p1, pn1h, KE);
    k_rownorm_f16_inplace<<<I_N, 256, 0, stream>>>(f0h, D0_N);
    k_rownorm_f16_inplace<<<I_N, 256, 0, stream>>>(f1h, D1_N);

    // fused sim + top-10 (WMMA), four graphs
    k_sim_topk<<<I_N / 16, 256, 0, stream>>>(pn0h, KE,   iL0, vL0);
    k_sim_topk<<<I_N / 16, 256, 0, stream>>>(pn1h, KE,   iL1, vL1);
    k_sim_topk<<<I_N / 16, 256, 0, stream>>>(f0h,  D0_N, iO0, vO0);
    k_sim_topk<<<I_N / 16, 256, 0, stream>>>(f1h,  D1_N, iO1, vO1);

    // item-graph laplacians + one LGConv layer
    k_item_dinv<<<cdiv(I_N, 256), 256, 0, stream>>>(iL0, vL0, iL1, vL1, iO0, vO0, iO1, vO1,
                                                    sm_w, dinv_l, dinv_o0, dinv_o1);
    k_item_spmm<<<cdiv((long long)I_N * KE, 256), 256, 0, stream>>>(gi, iL0, vL0, iL1, vL1,
                                                                    iO0, vO0, iO1, vO1, sm_w,
                                                                    dinv_l, dinv_o0, dinv_o1, item_emb);

    // UI graph degrees (norm = 1.0) : deg = incidence + 1 (diag), dinv = 1/deg
    k_fill_f32<<<cdiv(NN_N, 256), 256, 0, stream>>>(dinv_a, 1.0f, NN_N);
    k_ui_deg<<<cdiv(E, 256), 256, 0, stream>>>(ui_src, ui_dst, dinv_a, E);
    k_recip<<<cdiv(NN_N, 256), 256, 0, stream>>>(dinv_a, NN_N);

    // ego embeddings and two propagation layers with per-row l2norm
    k_concat_ego<<<cdiv((long long)NN_N * KE, 256), 256, 0, stream>>>(gu, gi, e0);

    k_ui_diag<<<cdiv((long long)NN_N * KE, 256), 256, 0, stream>>>(dinv_a, e0, e1);
    k_ui_edges<<<cdiv((long long)E * KE, 256), 256, 0, stream>>>(ui_src, ui_dst, dinv_a, e0, e1, E);
    k_l2norm_rows<<<NN_N, 64, 0, stream>>>(e1);

    k_ui_diag<<<cdiv((long long)NN_N * KE, 256), 256, 0, stream>>>(dinv_a, e1, e2);
    k_ui_edges<<<cdiv((long long)E * KE, 256), 256, 0, stream>>>(ui_src, ui_dst, dinv_a, e1, e2, E);
    k_l2norm_rows<<<NN_N, 64, 0, stream>>>(e2);

    // outputs: gu_out = mean; gi_out = mean + l2norm(item_emb)
    k_final_u<<<cdiv((long long)U_N * KE, 256), 256, 0, stream>>>(e0, e1, e2, out);
    k_final_i<<<I_N, 64, 0, stream>>>(e0, e1, e2, item_emb, out);
}
